// CfdModel_40664750359028
// MI455X (gfx1250) — compile-verified, matrix operand order checked by reading, not run on hardware
//
#include <hip/hip_runtime.h>
#include <hip/hip_bf16.h>
#include <cstdint>

// ---------------------------------------------------------------------------
// MeshGraphNets-style GNN for MI455X (gfx1250, wave32).
// All GEMMs (output width 128) run as bf16 WMMA 16x16x32 with fp32 accum.
// Weights are pre-packed into WMMA B-fragment order and staged in LDS
// (320 KB/WGP) once per block; A tiles are double-buffered through LDS.
// fp32 master copies of node/edge latents carry the LN + residual chain.
// ---------------------------------------------------------------------------

typedef __bf16 bf16;
typedef __attribute__((ext_vector_type(16))) __bf16 v16bf;
typedef __attribute__((ext_vector_type(8)))  __bf16 v8bf;
typedef __attribute__((ext_vector_type(8)))  float  v8f;

#define N_NODES 50000
#define N_EDGES 200000
#define MP_STEPS 15

// ---------------------------------------------------------------------------
// Weight packing: fp32 [K x 128] row-major -> bf16 WMMA B-fragments.
// Fragment (kc, nt) holds B rows kc*32..+31, cols nt*16..+15.
// Per ISA 7.12.2 (16-bit operand layout): lane = {khalf(1b), ncol(4b)},
// register element j -> K = khalf*8 + j (j<8) or 16 + khalf*8 + (j-8).
// Packed linear: ((kc*8 + nt)*32 + lane)*16 + j   (512 elems per fragment).
// ---------------------------------------------------------------------------
__global__ void pack_w_kernel(const float* __restrict__ W, bf16* __restrict__ out,
                              int Kphys, int Kpad, int N) {
  const int ntiles = N >> 4;
  const int total  = (Kpad >> 5) * ntiles * 512;
  for (int i = blockIdx.x * blockDim.x + threadIdx.x; i < total;
       i += gridDim.x * blockDim.x) {
    int j    = i & 15;
    int lane = (i >> 4) & 31;
    int frag = i >> 9;
    int nt = frag % ntiles;
    int kc = frag / ntiles;
    int hf  = lane >> 4;
    int nlo = lane & 15;
    int kk = (j < 8) ? (hf * 8 + j) : (16 + hf * 8 + (j - 8));
    int k = kc * 32 + kk;
    int n = nt * 16 + nlo;
    float v = (k < Kphys) ? W[(size_t)k * N + n] : 0.0f;
    out[i] = (bf16)v;
  }
}

// ---------------------------------------------------------------------------
// Feature builders (normalized, zero-padded to K=32) -> bf16 rows.
// ---------------------------------------------------------------------------
__global__ void featurize_node_kernel(const int* __restrict__ ntype,
                                      const float* __restrict__ vel,
                                      const float* __restrict__ mean,
                                      const float* __restrict__ stdv,
                                      bf16* __restrict__ out, int n) {
  int i = blockIdx.x * blockDim.x + threadIdx.x;
  if (i >= n) return;
  float f[11];
  f[0] = vel[(size_t)i * 2 + 0];
  f[1] = vel[(size_t)i * 2 + 1];
  int t = ntype[i];
#pragma unroll
  for (int c = 0; c < 9; ++c) f[2 + c] = (c == t) ? 1.0f : 0.0f;
  bf16* o = out + (size_t)i * 32;
#pragma unroll
  for (int c = 0; c < 11; ++c) o[c] = (bf16)((f[c] - mean[c]) / stdv[c]);
#pragma unroll
  for (int c = 11; c < 32; ++c) o[c] = (bf16)0.0f;
}

__global__ void featurize_edge_kernel(const float* __restrict__ mesh,
                                      const int* __restrict__ srcs,
                                      const int* __restrict__ dsts,
                                      const float* __restrict__ mean,
                                      const float* __restrict__ stdv,
                                      bf16* __restrict__ out, int n) {
  int i = blockIdx.x * blockDim.x + threadIdx.x;
  if (i >= n) return;
  int s = srcs[i], d = dsts[i];
  float rx = mesh[(size_t)s * 2 + 0] - mesh[(size_t)d * 2 + 0];
  float ry = mesh[(size_t)s * 2 + 1] - mesh[(size_t)d * 2 + 1];
  float nr = sqrtf(rx * rx + ry * ry);
  bf16* o = out + (size_t)i * 32;
  o[0] = (bf16)((rx - mean[0]) / stdv[0]);
  o[1] = (bf16)((ry - mean[1]) / stdv[1]);
  o[2] = (bf16)((nr - mean[2]) / stdv[2]);
#pragma unroll
  for (int c = 3; c < 32; ++c) o[c] = (bf16)0.0f;
}

__global__ void zero_f32_kernel(float* __restrict__ p, long long n) {
  long long i = (long long)blockIdx.x * blockDim.x + threadIdx.x;
  long long st = (long long)gridDim.x * blockDim.x;
  for (; i < n; i += st) p[i] = 0.0f;
}

// ---------------------------------------------------------------------------
// Fused GEMM: Y[rows x 128] = act(A[rows x K] * W + b) with epilogues.
// K is a template parameter (32/128/256/384).
// IN_MODE: 0 = bf16 row-major A, 1 = [e | n[src] | n[dst]] fp32 gather-concat,
//          2 = [n | agg] fp32 concat.
// EPI: 0 = bias+ReLU -> bf16, 1 = bias+LN -> f32,
//      2 = bias+LN+res -> f32 + atomic scatter into agg[dst],
//      3 = bias+LN+res -> f32 + zero agg row (prep next step).
// Block: 256 thr (8 waves), tile 128 rows x 128 cols; wave = 16 rows.
// Dynamic LDS: [ packed W : K*256 B ][ A tile double buffer : 2*8 KB ].
// Full tiles (all but the last block) take branch-free store paths.
// ---------------------------------------------------------------------------
struct GemmArgs {
  const bf16*  A;        // IN_MODE 0
  int          ldA;
  const float* seg0;     // e (mode1) or n (mode2)
  const float* nsrc;     // node table for gathers (mode1)
  const float* seg1;     // agg (mode2)
  const int*   srcs;
  const int*   dsts;
  const bf16*  Wp;       // packed weights
  const float* bias;
  const float* ln_g;
  const float* ln_b;
  const float* res;
  float*       out_f32;
  bf16*        out_b16;
  float*       agg;      // scatter target (EPI 2) / zero target (EPI 3)
  int          nrows;
};

template <int IN_MODE>
__device__ __forceinline__ void stage_chunk(const GemmArgs& g, int kc, int grow,
                                            int scol, bf16* __restrict__ sdst,
                                            bool inb) {
  if constexpr (IN_MODE == 0) {
    if (inb) {
      const bf16* src = g.A + (size_t)grow * g.ldA + kc * 32 + scol;
      ((v8bf*)sdst)[0] = ((const v8bf*)src)[0];
      ((v8bf*)sdst)[1] = ((const v8bf*)src)[1];
    } else {
#pragma unroll
      for (int q = 0; q < 16; ++q) sdst[q] = (bf16)0.0f;
    }
  } else {
    const float* src = nullptr;
    if (inb) {
      const int seg = kc >> 2;
      const int off = (kc & 3) * 32 + scol;
      if constexpr (IN_MODE == 1) {
        if (seg == 0)      src = g.seg0 + (size_t)grow * 128 + off;
        else if (seg == 1) src = g.nsrc + (size_t)g.srcs[grow] * 128 + off;
        else               src = g.nsrc + (size_t)g.dsts[grow] * 128 + off;
      } else {
        src = (seg == 0) ? (g.seg0 + (size_t)grow * 128 + off)
                         : (g.seg1 + (size_t)grow * 128 + off);
      }
    }
    if (src) {
#pragma unroll
      for (int q = 0; q < 4; ++q) {
        float4 f = ((const float4*)src)[q];
        sdst[q * 4 + 0] = (bf16)f.x;
        sdst[q * 4 + 1] = (bf16)f.y;
        sdst[q * 4 + 2] = (bf16)f.z;
        sdst[q * 4 + 3] = (bf16)f.w;
      }
    } else {
#pragma unroll
      for (int q = 0; q < 16; ++q) sdst[q] = (bf16)0.0f;
    }
  }
}

template <int K, int IN_MODE, int EPI>
__launch_bounds__(256)
__global__ void gemm128_kernel(GemmArgs g) {
  constexpr int NKC = K / 32;
  extern __shared__ char smem[];
  bf16* lds_w = (bf16*)smem;                        // K*128 bf16
  bf16* lds_a = (bf16*)(smem + (size_t)K * 256);    // 2 x (128*32) bf16

  const int tid  = threadIdx.x;
  const int wave = tid >> 5;
  const int lane = tid & 31;
  const int tileRow = blockIdx.x * 128;

  // ---- cooperative copy: packed weights -> LDS (once per block) ----
  {
    const uint4* src = (const uint4*)g.Wp;
    uint4* dst = (uint4*)lds_w;
    constexpr int NV = (K * 128) / 8;  // 16-byte vectors; multiple of 256
#pragma unroll
    for (int i = 0; i < NV / 256; ++i) dst[tid + i * 256] = src[tid + i * 256];
  }

  v8f acc[8];
#pragma unroll
  for (int t = 0; t < 8; ++t)
#pragma unroll
    for (int r = 0; r < 8; ++r) acc[t][r] = 0.0f;

  const int srow = tid >> 1;         // staging row 0..127
  const int scol = (tid & 1) * 16;   // staging cols {0,16}
  const int grow = tileRow + srow;
  const int soff = srow * 32 + scol;
  const bool inb = grow < g.nrows;   // loop-invariant bounds check

  // prime buffer 0 with chunk 0
  stage_chunk<IN_MODE>(g, 0, grow, scol, lds_a + soff, inb);

#pragma unroll
  for (int kc = 0; kc < NKC; ++kc) {
    __syncthreads();
    if (kc + 1 < NKC)
      stage_chunk<IN_MODE>(g, kc + 1, grow, scol,
                           lds_a + ((kc + 1) & 1) * (128 * 32) + soff, inb);

    // ---- A fragment: wave's 16 rows, per ISA 16-bit operand layout ----
    const int hf = lane >> 4;
    const bf16* arow =
        lds_a + (kc & 1) * (128 * 32) + (wave * 16 + (lane & 15)) * 32;
    v8bf alo = *(const v8bf*)(arow + hf * 8);
    v8bf ahi = *(const v8bf*)(arow + 16 + hf * 8);
    v16bf a;
#pragma unroll
    for (int j = 0; j < 8; ++j) { a[j] = alo[j]; a[8 + j] = ahi[j]; }

    // ---- 8 WMMAs across the 128 output columns, B from LDS ----
    const bf16* wrow = lds_w + kc * (8 * 512) + lane * 16;
#pragma unroll
    for (int nt = 0; nt < 8; ++nt) {
      v8bf blo = *(const v8bf*)(wrow + nt * 512);
      v8bf bhi = *(const v8bf*)(wrow + nt * 512 + 8);
      v16bf b;
#pragma unroll
      for (int j = 0; j < 8; ++j) { b[j] = blo[j]; b[8 + j] = bhi[j]; }
      acc[nt] = __builtin_amdgcn_wmma_f32_16x16x32_bf16(
          false, a, false, b, (short)0, acc[nt], false, false);
    }
  }

  // ---- epilogue (C/D layout: lane={half,ncol}, reg r -> row half*8+r) ----
  const int hf  = lane >> 4;
  const int nlo = lane & 15;
  const int rowBase = tileRow + wave * 16 + hf * 8;
  const bool full = (tileRow + 128) <= g.nrows;  // block-uniform fast path

  if constexpr (EPI == 0) {  // bias + ReLU -> bf16
    float bv[8];
#pragma unroll
    for (int nt = 0; nt < 8; ++nt) bv[nt] = g.bias[nt * 16 + nlo];
    bf16* outp = g.out_b16 + (size_t)rowBase * 128 + nlo;
    if (full) {
#pragma unroll
      for (int nt = 0; nt < 8; ++nt)
#pragma unroll
        for (int r = 0; r < 8; ++r) {
          float v = acc[nt][r] + bv[nt];
          v = v > 0.0f ? v : 0.0f;
          outp[r * 128 + nt * 16] = (bf16)v;
        }
    } else {
#pragma unroll
      for (int nt = 0; nt < 8; ++nt)
#pragma unroll
        for (int r = 0; r < 8; ++r) {
          if (rowBase + r < g.nrows) {
            float v = acc[nt][r] + bv[nt];
            v = v > 0.0f ? v : 0.0f;
            outp[r * 128 + nt * 16] = (bf16)v;
          }
        }
    }
  } else {  // bias + LayerNorm (+residual) (+scatter/zero) -> f32
    float lgv[8], lbv[8];
#pragma unroll
    for (int nt = 0; nt < 8; ++nt) {
      const float bcol = g.bias[nt * 16 + nlo];
      lgv[nt] = g.ln_g[nt * 16 + nlo];
      lbv[nt] = g.ln_b[nt * 16 + nlo];
#pragma unroll
      for (int r = 0; r < 8; ++r) acc[nt][r] += bcol;
    }
    float mean_[8], rstd_[8];
#pragma unroll
    for (int r = 0; r < 8; ++r) {
      float s = 0.0f, q = 0.0f;
#pragma unroll
      for (int nt = 0; nt < 8; ++nt) {
        const float v = acc[nt][r];
        s += v;
        q += v * v;
      }
#pragma unroll
      for (int off = 1; off < 16; off <<= 1) {  // reduce within 16-lane half
        s += __shfl_xor(s, off, 16);
        q += __shfl_xor(q, off, 16);
      }
      mean_[r] = s * (1.0f / 128.0f);
      const float var = q * (1.0f / 128.0f) - mean_[r] * mean_[r];
      rstd_[r] = rsqrtf(var + 1e-5f);
    }
    float* outp = g.out_f32 + (size_t)rowBase * 128 + nlo;
    const float* resp =
        (EPI >= 2) ? (g.res + (size_t)rowBase * 128 + nlo) : nullptr;
    float* aggzp =
        (EPI == 3) ? (g.agg + (size_t)rowBase * 128 + nlo) : nullptr;
    if (full) {
#pragma unroll
      for (int r = 0; r < 8; ++r) {
        int d = 0;
        if constexpr (EPI == 2) d = g.dsts[rowBase + r];
#pragma unroll
        for (int nt = 0; nt < 8; ++nt) {
          float v = (acc[nt][r] - mean_[r]) * rstd_[r] * lgv[nt] + lbv[nt];
          if constexpr (EPI >= 2) v += resp[r * 128 + nt * 16];
          outp[r * 128 + nt * 16] = v;
          if constexpr (EPI == 2)
            atomicAdd(&g.agg[(size_t)d * 128 + nt * 16 + nlo], v);
          if constexpr (EPI == 3) aggzp[r * 128 + nt * 16] = 0.0f;
        }
      }
    } else {
#pragma unroll
      for (int r = 0; r < 8; ++r) {
        if (rowBase + r < g.nrows) {
          int d = 0;
          if constexpr (EPI == 2) d = g.dsts[rowBase + r];
#pragma unroll
          for (int nt = 0; nt < 8; ++nt) {
            float v = (acc[nt][r] - mean_[r]) * rstd_[r] * lgv[nt] + lbv[nt];
            if constexpr (EPI >= 2) v += resp[r * 128 + nt * 16];
            outp[r * 128 + nt * 16] = v;
            if constexpr (EPI == 2)
              atomicAdd(&g.agg[(size_t)d * 128 + nt * 16 + nlo], v);
            if constexpr (EPI == 3) aggzp[r * 128 + nt * 16] = 0.0f;
          }
        }
      }
    }
  }
}

// Decoder last layer (128 -> 2) + output denorm; tiny, plain VALU.
__global__ void dec_final_kernel(const bf16* __restrict__ h,
                                 const float* __restrict__ W2,
                                 const float* __restrict__ b2,
                                 const float* __restrict__ om,
                                 const float* __restrict__ os,
                                 float* __restrict__ out, int n) {
  int i = blockIdx.x * blockDim.x + threadIdx.x;
  if (i >= n) return;
  float a0 = b2[0], a1 = b2[1];
  const bf16* hr = h + (size_t)i * 128;
#pragma unroll 8
  for (int k = 0; k < 128; ++k) {
    const float hv = (float)hr[k];
    a0 += hv * W2[k * 2 + 0];
    a1 += hv * W2[k * 2 + 1];
  }
  out[(size_t)i * 2 + 0] = a0 * os[0] + om[0];
  out[(size_t)i * 2 + 1] = a1 * os[1] + om[1];
}

// ---------------------------------------------------------------------------
// Host side
// ---------------------------------------------------------------------------
enum {
  I_NODE_TYPE = 0, I_VEL, I_MESH, I_SRCS, I_DSTS,
  I_NMEAN, I_NSTD, I_EMEAN, I_ESTD, I_OMEAN, I_OSTD,
  I_ENN_W0, I_ENN_W1, I_ENN_W2, I_ENN_B0, I_ENN_B1, I_ENN_B2, I_ENN_G, I_ENN_BT,
  I_ENE_W0, I_ENE_W1, I_ENE_W2, I_ENE_B0, I_ENE_B1, I_ENE_B2, I_ENE_G, I_ENE_BT,
  I_PE_W0, I_PE_W1, I_PE_W2, I_PE_B0, I_PE_B1, I_PE_B2, I_PE_G, I_PE_BT,
  I_PN_W0, I_PN_W1, I_PN_W2, I_PN_B0, I_PN_B1, I_PN_B2, I_PN_G, I_PN_BT,
  I_DEC_W0, I_DEC_W1, I_DEC_W2, I_DEC_B0, I_DEC_B1, I_DEC_B2
};

template <int K, int IN, int EPI>
static void launch_one(const GemmArgs& g, int grid, hipStream_t s) {
  const size_t shmem = (size_t)K * 256 + 2 * 128 * 32 * sizeof(bf16);
  (void)hipFuncSetAttribute(
      reinterpret_cast<const void*>(&gemm128_kernel<K, IN, EPI>),
      hipFuncAttributeMaxDynamicSharedMemorySize, (int)shmem);
  gemm128_kernel<K, IN, EPI><<<grid, 256, shmem, s>>>(g);
}

extern "C" void kernel_launch(void* const* d_in, const int* in_sizes, int n_in,
                              void* d_out, int out_size, void* d_ws,
                              size_t ws_size, hipStream_t stream) {
  (void)in_sizes; (void)n_in; (void)out_size; (void)ws_size;

  const int*   node_type = (const int*)d_in[I_NODE_TYPE];
  const float* velocity  = (const float*)d_in[I_VEL];
  const float* mesh_pos  = (const float*)d_in[I_MESH];
  const int*   srcs      = (const int*)d_in[I_SRCS];
  const int*   dsts      = (const int*)d_in[I_DSTS];

  // ---- workspace carve (needs ~226 MB) ----
  char* ws = (char*)d_ws;
  size_t off = 0;
  auto carve = [&](size_t bytes) {
    size_t o = off;
    off = (off + bytes + 255) & ~(size_t)255;
    return o;
  };
  float* e_buf   = (float*)(ws + carve((size_t)N_EDGES * 128 * 4));
  float* n_buf   = (float*)(ws + carve((size_t)N_NODES * 128 * 4));
  float* agg_buf = (float*)(ws + carve((size_t)N_NODES * 128 * 4));
  bf16*  h_buf   = (bf16*)(ws + carve((size_t)N_EDGES * 128 * 2));
  bf16*  fe_buf  = (bf16*)(ws + carve((size_t)N_EDGES * 32 * 2));
  bf16*  fn_buf  = (bf16*)(ws + carve((size_t)N_NODES * 32 * 2));
  // packed weight element counts: Kpad * 128 each
  bf16* pw = (bf16*)(ws + carve((size_t)2400000 * 2));
  size_t po = 0;
  auto carve_pw = [&](size_t elems) { size_t o = po; po += elems; return o; };
  bf16* enn_w0p = pw + carve_pw(32 * 128);
  bf16* enn_w1p = pw + carve_pw(128 * 128);
  bf16* enn_w2p = pw + carve_pw(128 * 128);
  bf16* ene_w0p = pw + carve_pw(32 * 128);
  bf16* ene_w1p = pw + carve_pw(128 * 128);
  bf16* ene_w2p = pw + carve_pw(128 * 128);
  bf16* dec_w0p = pw + carve_pw(128 * 128);
  bf16* dec_w1p = pw + carve_pw(128 * 128);
  bf16* pe_w0p  = pw + carve_pw((size_t)MP_STEPS * 384 * 128);
  bf16* pe_w1p  = pw + carve_pw((size_t)MP_STEPS * 128 * 128);
  bf16* pe_w2p  = pw + carve_pw((size_t)MP_STEPS * 128 * 128);
  bf16* pn_w0p  = pw + carve_pw((size_t)MP_STEPS * 256 * 128);
  bf16* pn_w1p  = pw + carve_pw((size_t)MP_STEPS * 128 * 128);
  bf16* pn_w2p  = pw + carve_pw((size_t)MP_STEPS * 128 * 128);

  auto pack = [&](const float* W, bf16* dst, int Kphys, int Kpad) {
    int total = Kpad * 128;
    pack_w_kernel<<<(total + 255) / 256, 256, 0, stream>>>(W, dst, Kphys, Kpad, 128);
  };
  pack((const float*)d_in[I_ENN_W0], enn_w0p, 11, 32);
  pack((const float*)d_in[I_ENN_W1], enn_w1p, 128, 128);
  pack((const float*)d_in[I_ENN_W2], enn_w2p, 128, 128);
  pack((const float*)d_in[I_ENE_W0], ene_w0p, 3, 32);
  pack((const float*)d_in[I_ENE_W1], ene_w1p, 128, 128);
  pack((const float*)d_in[I_ENE_W2], ene_w2p, 128, 128);
  pack((const float*)d_in[I_DEC_W0], dec_w0p, 128, 128);
  pack((const float*)d_in[I_DEC_W1], dec_w1p, 128, 128);
  for (int s = 0; s < MP_STEPS; ++s) {
    pack((const float*)d_in[I_PE_W0] + (size_t)s * 384 * 128, pe_w0p + (size_t)s * 384 * 128, 384, 384);
    pack((const float*)d_in[I_PE_W1] + (size_t)s * 128 * 128, pe_w1p + (size_t)s * 128 * 128, 128, 128);
    pack((const float*)d_in[I_PE_W2] + (size_t)s * 128 * 128, pe_w2p + (size_t)s * 128 * 128, 128, 128);
    pack((const float*)d_in[I_PN_W0] + (size_t)s * 256 * 128, pn_w0p + (size_t)s * 256 * 128, 256, 256);
    pack((const float*)d_in[I_PN_W1] + (size_t)s * 128 * 128, pn_w1p + (size_t)s * 128 * 128, 128, 128);
    pack((const float*)d_in[I_PN_W2] + (size_t)s * 128 * 128, pn_w2p + (size_t)s * 128 * 128, 128, 128);
  }

  // ---- features + agg init ----
  featurize_node_kernel<<<(N_NODES + 255) / 256, 256, 0, stream>>>(
      node_type, velocity, (const float*)d_in[I_NMEAN], (const float*)d_in[I_NSTD],
      fn_buf, N_NODES);
  featurize_edge_kernel<<<(N_EDGES + 255) / 256, 256, 0, stream>>>(
      mesh_pos, srcs, dsts, (const float*)d_in[I_EMEAN], (const float*)d_in[I_ESTD],
      fe_buf, N_EDGES);
  zero_f32_kernel<<<1024, 256, 0, stream>>>(agg_buf, (long long)N_NODES * 128);

  const int gridN = (N_NODES + 127) / 128;
  const int gridE = (N_EDGES + 127) / 128;

  // ---- encoders ----
  {  // node encoder -> n_buf
    GemmArgs g{}; g.A = fn_buf; g.ldA = 32; g.Wp = enn_w0p;
    g.bias = (const float*)d_in[I_ENN_B0]; g.out_b16 = h_buf; g.nrows = N_NODES;
    launch_one<32, 0, 0>(g, gridN, stream);
    g = GemmArgs{}; g.A = h_buf; g.ldA = 128; g.Wp = enn_w1p;
    g.bias = (const float*)d_in[I_ENN_B1]; g.out_b16 = h_buf; g.nrows = N_NODES;
    launch_one<128, 0, 0>(g, gridN, stream);
    g = GemmArgs{}; g.A = h_buf; g.ldA = 128; g.Wp = enn_w2p;
    g.bias = (const float*)d_in[I_ENN_B2];
    g.ln_g = (const float*)d_in[I_ENN_G]; g.ln_b = (const float*)d_in[I_ENN_BT];
    g.out_f32 = n_buf; g.nrows = N_NODES;
    launch_one<128, 0, 1>(g, gridN, stream);
  }
  {  // edge encoder -> e_buf
    GemmArgs g{}; g.A = fe_buf; g.ldA = 32; g.Wp = ene_w0p;
    g.bias = (const float*)d_in[I_ENE_B0]; g.out_b16 = h_buf; g.nrows = N_EDGES;
    launch_one<32, 0, 0>(g, gridE, stream);
    g = GemmArgs{}; g.A = h_buf; g.ldA = 128; g.Wp = ene_w1p;
    g.bias = (const float*)d_in[I_ENE_B1]; g.out_b16 = h_buf; g.nrows = N_EDGES;
    launch_one<128, 0, 0>(g, gridE, stream);
    g = GemmArgs{}; g.A = h_buf; g.ldA = 128; g.Wp = ene_w2p;
    g.bias = (const float*)d_in[I_ENE_B2];
    g.ln_g = (const float*)d_in[I_ENE_G]; g.ln_b = (const float*)d_in[I_ENE_BT];
    g.out_f32 = e_buf; g.nrows = N_EDGES;
    launch_one<128, 0, 1>(g, gridE, stream);
  }

  // ---- 15 message-passing steps ----
  for (int s = 0; s < MP_STEPS; ++s) {
    // edge L1: [e | n[src] | n[dst]] (K=384) -> ReLU h
    GemmArgs g{}; g.seg0 = e_buf; g.nsrc = n_buf; g.srcs = srcs; g.dsts = dsts;
    g.Wp = pe_w0p + (size_t)s * 384 * 128;
    g.bias = (const float*)d_in[I_PE_B0] + s * 128;
    g.out_b16 = h_buf; g.nrows = N_EDGES;
    launch_one<384, 1, 0>(g, gridE, stream);
    // edge L2
    g = GemmArgs{}; g.A = h_buf; g.ldA = 128;
    g.Wp = pe_w1p + (size_t)s * 128 * 128;
    g.bias = (const float*)d_in[I_PE_B1] + s * 128;
    g.out_b16 = h_buf; g.nrows = N_EDGES;
    launch_one<128, 0, 0>(g, gridE, stream);
    // edge L3: LN + residual(e) -> e, atomic scatter-add into agg[dst]
    g = GemmArgs{}; g.A = h_buf; g.ldA = 128;
    g.Wp = pe_w2p + (size_t)s * 128 * 128;
    g.bias = (const float*)d_in[I_PE_B2] + s * 128;
    g.ln_g = (const float*)d_in[I_PE_G] + s * 128;
    g.ln_b = (const float*)d_in[I_PE_BT] + s * 128;
    g.res = e_buf; g.out_f32 = e_buf; g.agg = agg_buf; g.dsts = dsts;
    g.nrows = N_EDGES;
    launch_one<128, 0, 2>(g, gridE, stream);
    // node L1: [n | agg] (K=256) -> ReLU h
    g = GemmArgs{}; g.seg0 = n_buf; g.seg1 = agg_buf;
    g.Wp = pn_w0p + (size_t)s * 256 * 128;
    g.bias = (const float*)d_in[I_PN_B0] + s * 128;
    g.out_b16 = h_buf; g.nrows = N_NODES;
    launch_one<256, 2, 0>(g, gridN, stream);
    // node L2
    g = GemmArgs{}; g.A = h_buf; g.ldA = 128;
    g.Wp = pn_w1p + (size_t)s * 128 * 128;
    g.bias = (const float*)d_in[I_PN_B1] + s * 128;
    g.out_b16 = h_buf; g.nrows = N_NODES;
    launch_one<128, 0, 0>(g, gridN, stream);
    // node L3: LN + residual(n) -> n, zero agg rows for next step
    g = GemmArgs{}; g.A = h_buf; g.ldA = 128;
    g.Wp = pn_w2p + (size_t)s * 128 * 128;
    g.bias = (const float*)d_in[I_PN_B2] + s * 128;
    g.ln_g = (const float*)d_in[I_PN_G] + s * 128;
    g.ln_b = (const float*)d_in[I_PN_BT] + s * 128;
    g.res = n_buf; g.out_f32 = n_buf; g.agg = agg_buf;
    g.nrows = N_NODES;
    launch_one<128, 0, 3>(g, gridN, stream);
  }

  // ---- decoder ----
  {
    GemmArgs g{}; g.seg0 = n_buf; g.seg1 = agg_buf;  // K=128 -> reads n only
    g.Wp = dec_w0p; g.bias = (const float*)d_in[I_DEC_B0];
    g.out_b16 = h_buf; g.nrows = N_NODES;
    launch_one<128, 2, 0>(g, gridN, stream);
    g = GemmArgs{}; g.A = h_buf; g.ldA = 128; g.Wp = dec_w1p;
    g.bias = (const float*)d_in[I_DEC_B1]; g.out_b16 = h_buf; g.nrows = N_NODES;
    launch_one<128, 0, 0>(g, gridN, stream);
    dec_final_kernel<<<(N_NODES + 255) / 256, 256, 0, stream>>>(
        h_buf, (const float*)d_in[I_DEC_W2], (const float*)d_in[I_DEC_B2],
        (const float*)d_in[I_OMEAN], (const float*)d_in[I_OSTD],
        (float*)d_out, N_NODES);
  }
}
